// FourierCrossAttention_84164179133301
// MI455X (gfx1250) — matrix operand order repaired and checked by reading, not compiled, not run on hardware
//
#include <hip/hip_runtime.h>
#include <math.h>

// ---------------------------------------------------------------------------
// FourierCrossAttention for MI455X (gfx1250, wave32, WMMA bf16)
// B=16, L=S=1024, D=1024, H=16, DK=64, MODES=64
// All stages are bf16 WMMA GEMMs (fp32 accumulate):
//   1) Q/K/V projections   : 16384x1024x1024 (x3)   [double-buffered pipeline]
//   2) truncated rDFT      : per (t,b,h): [cos;-sin](128x1024) @ X(1024x64)
//   3) spectral attention  : per (b,h): 64x64x128 + softmax + 64x64x64 (x2)
//   4) truncated irDFT     : per (b,h): G(1024x128) @ [Or;Oi](128x64)
//   5) output projection   : 16384x1024x1024        [double-buffered pipeline]
// Async global->LDS (ASYNCcnt) used for bulk bf16 staging where available.
// ---------------------------------------------------------------------------

typedef __attribute__((ext_vector_type(16))) __bf16 v16bf;
typedef __attribute__((ext_vector_type(8)))  float  v8f;

#define B_  16
#define L_  1024
#define D_  1024
#define H_  16
#define DK_ 64
#define BL_ (B_ * L_)   // 16384
#define BH_ (B_ * H_)   // 256

// ---- gfx1250 async global->LDS path (guarded: falls back to vector copy) ---
#if defined(__HIP_DEVICE_COMPILE__) && __has_builtin(__builtin_amdgcn_global_load_async_to_lds_b128)
#define USE_ASYNC_LDS 1
#else
#define USE_ASYNC_LDS 0
#endif

typedef __attribute__((__vector_size__(4 * sizeof(int)))) int vi4;

__device__ __forceinline__ void cp16_g2l(const __bf16* g, __bf16* l) {
#if USE_ASYNC_LDS
  __builtin_amdgcn_global_load_async_to_lds_b128(
      (__attribute__((address_space(1))) vi4*)g,
      (__attribute__((address_space(3))) vi4*)l, 0, 0);
#else
  *(uint4*)l = *(const uint4*)g;
#endif
}

__device__ __forceinline__ void async_fence() {
#if USE_ASYNC_LDS
#if __has_builtin(__builtin_amdgcn_s_wait_asynccnt)
  __builtin_amdgcn_s_wait_asynccnt(0);
#else
  asm volatile("s_wait_asynccnt 0x0" ::: "memory");
#endif
#endif
}

// ---------------------------------------------------------------------------
__device__ __forceinline__ v8f wmma_bf16(v16bf a, v16bf b, v8f c) {
  return __builtin_amdgcn_wmma_f32_16x16x32_bf16(false, a, false, b, (short)0, c,
                                                 false, false);
}

__device__ __forceinline__ v8f zero8() {
  v8f z = {0.f, 0.f, 0.f, 0.f, 0.f, 0.f, 0.f, 0.f};
  return z;
}

// A fragment: 16x32 tile, row-major LDS, p -> (row0, k0). Lane r holds row M=r%16;
// lanes 0-15: K 0..15; lanes 16-31: K 16..31.
__device__ __forceinline__ v16bf frag_a(const __bf16* p, int lda) {
  const int lane = threadIdx.x & 31;
  const int r  = lane & 15;
  const int kh = (lane >> 4) << 4;            // 0 or 16
  v16bf f;
#pragma unroll
  for (int e = 0; e < 16; ++e) f[e] = p[r * lda + kh + e];
  return f;
}

// B fragment from N-major (transposed) LDS tile: p -> row n0 of [N][ldk], col k0.
__device__ __forceinline__ v16bf frag_b_t(const __bf16* p, int ldk) {
  const int lane = threadIdx.x & 31;
  const int n    = lane & 15;
  const int koff = (lane >> 4) << 4;          // 0 or 16
  v16bf f;
#pragma unroll
  for (int e = 0; e < 16; ++e) f[e] = p[n * ldk + koff + e];
  return f;
}

// B fragment from row-major [K][N] LDS (strided per-lane loads).
__device__ __forceinline__ v16bf frag_b_rm(const __bf16* p, int ldn) {
  const int lane = threadIdx.x & 31;
  const int n    = lane & 15;
  const int koff = (lane >> 4) << 4;
  v16bf f;
#pragma unroll
  for (int e = 0; e < 16; ++e) f[e] = p[(koff + e) * ldn + n];
  return f;
}

// C/D 16x16 f32 accumulator store: lane n holds col N=n%16; VGPR v holds row
// M = v + (lane<16 ? 0 : 8).
template <typename TC>
__device__ __forceinline__ void store_acc(TC* c, int ldc, v8f acc) {
  const int lane = threadIdx.x & 31;
  const int n    = lane & 15;
  const int moff = (lane >> 4) << 3;
#pragma unroll
  for (int v = 0; v < 8; ++v) c[(size_t)(moff + v) * ldc + n] = (TC)acc[v];
}

// ---------------------------------------------------------------------------
// DFT basis builders (exact phase reduction: m*l mod 1024 keeps trig args small)
// F (128x1024): rows 0..63 cos(2*pi*m*l/N), rows 64..127 -sin(...)   [rfft]
// G (1024x128): cols 0..63 (m==0?1:2cos)/N, cols 64..127 (m==0?0:-2sin)/N
// ---------------------------------------------------------------------------
__global__ void fill_F_kernel(__bf16* __restrict__ F) {
  int i = blockIdx.x * 256 + threadIdx.x;
  if (i >= 128 * 1024) return;
  int m = i >> 10, l = i & 1023;
  int ph = ((m & 63) * l) & 1023;
  float th = (float)ph * (6.28318530717958647692f / 1024.0f);
  F[i] = (__bf16)((m < 64) ? cosf(th) : -sinf(th));
}

__global__ void fill_G_kernel(__bf16* __restrict__ G) {
  int i = blockIdx.x * 256 + threadIdx.x;
  if (i >= 1024 * 128) return;
  int l = i >> 7, c = i & 127, m = c & 63;
  int ph = (m * l) & 1023;
  float th = (float)ph * (6.28318530717958647692f / 1024.0f);
  float v;
  if (c < 64) v = (m == 0 ? 1.0f : 2.0f * cosf(th)) * (1.0f / 1024.0f);
  else        v = (m == 0 ? 0.0f : -2.0f * sinf(th)) * (1.0f / 1024.0f);
  G[i] = (__bf16)v;
}

// ---------------------------------------------------------------------------
// Big GEMM: C(MxN) = A(MxK) @ B(KxN) + bias.  A: TA row-major, B: f32 row-major
// (converted to bf16 while staging), C: TC.  Block tile 128x128, BK=32,
// 8 waves arranged 4(M) x 2(N); each wave computes 32x64 = 8 WMMA tiles.
// Double-buffered LDS with register staging: next tile's global loads are in
// flight while the current tile's WMMAs issue; one barrier per K-step.
// ---------------------------------------------------------------------------
template <typename TA, typename TC>
__global__ __launch_bounds__(256) void gemm_bias_kernel(
    const TA* __restrict__ A, const float* __restrict__ Bm,
    const float* __restrict__ bias, TC* __restrict__ C,
    int M, int N, int K) {
  const int LD = 40;                                  // padded K-extent in LDS
  __shared__ __align__(16) __bf16 lA[2][128 * LD];    // [buf][row][k]
  __shared__ __align__(16) __bf16 lBt[2][128 * LD];   // [buf][n][k]
  const int bm = blockIdx.y * 128;
  const int bn = blockIdx.x * 128;
  const int tid = threadIdx.x;
  const int w  = tid >> 5;
  const int wm = (w >> 1) * 32;
  const int wn = (w & 1) * 64;
  // staging coordinates
  const int ar = tid >> 1, ac = (tid & 1) * 16;   // A: 128x32, 16 elems/thread
  const int bk = tid >> 3, bn2 = (tid & 7) * 16;  // B: 32x128, 16 elems/thread

  v8f acc[2][4];
#pragma unroll
  for (int i = 0; i < 2; ++i)
#pragma unroll
    for (int j = 0; j < 4; ++j) acc[i][j] = zero8();

  float ra[16], rb[16];
  auto load_tile = [&](int k0) {
    const TA* ga = A + (size_t)(bm + ar) * K + k0 + ac;
#pragma unroll
    for (int e = 0; e < 16; ++e) ra[e] = (float)ga[e];
    const float* gb = Bm + (size_t)(k0 + bk) * N + bn + bn2;
#pragma unroll
    for (int e = 0; e < 16; ++e) rb[e] = gb[e];
  };
  auto store_tile = [&](int buf) {
#pragma unroll
    for (int e = 0; e < 16; ++e) lA[buf][ar * LD + ac + e] = (__bf16)ra[e];
#pragma unroll
    for (int e = 0; e < 16; ++e) lBt[buf][(bn2 + e) * LD + bk] = (__bf16)rb[e];
  };

  load_tile(0);
  store_tile(0);
  __syncthreads();

  const int nk = K / 32;
  for (int t = 0; t < nk; ++t) {
    const int cur = t & 1;
    if (t + 1 < nk) load_tile((t + 1) * 32);   // global loads in flight
#pragma unroll
    for (int i = 0; i < 2; ++i) {
      v16bf af = frag_a(&lA[cur][(wm + i * 16) * LD], LD);
#pragma unroll
      for (int j = 0; j < 4; ++j) {
        v16bf bf = frag_b_t(&lBt[cur][(wn + j * 16) * LD], LD);
        acc[i][j] = wmma_bf16(af, bf, acc[i][j]);
      }
    }
    if (t + 1 < nk) store_tile(cur ^ 1);       // deposit into idle buffer
    __syncthreads();
  }

  const int lane = tid & 31;
  const int n = lane & 15;
#pragma unroll
  for (int i = 0; i < 2; ++i)
#pragma unroll
    for (int j = 0; j < 4; ++j) {
      float bv = bias[bn + wn + j * 16 + n];
      v8f a2 = acc[i][j];
#pragma unroll
      for (int v = 0; v < 8; ++v) a2[v] += bv;
      TC* cp = C + (size_t)(bm + wm + i * 16) * N + bn + wn + j * 16;
      store_acc<TC>(cp, N, a2);
    }
}

// ---------------------------------------------------------------------------
// Truncated rDFT: per (t,bh): ft(128x64) = F(128x1024) @ X(1024x64)
// X is a strided view into proj_t: X[l,dk] = proj_t[(b*L+l)*D + h*64+dk]
// F tile staged with async global->LDS; X tile transposed via ds stores.
// ---------------------------------------------------------------------------
__global__ __launch_bounds__(256) void dft_kernel(
    const __bf16* __restrict__ F, const __bf16* __restrict__ proj,
    __bf16* __restrict__ ft) {
  const int LD = 40;
  __shared__ __align__(16) __bf16 lF[128 * LD];
  __shared__ __align__(16) __bf16 lXt[64 * LD];
  const int t = blockIdx.x / BH_;
  const int bh = blockIdx.x % BH_;
  const int b = bh >> 4, h = bh & 15;
  const __bf16* X = proj + (size_t)t * BL_ * D_ + (size_t)b * L_ * D_ + h * 64;
  __bf16* out = ft + (size_t)t * BH_ * 128 * 64 + (size_t)bh * 128 * 64;
  const int tid = threadIdx.x, w = tid >> 5;

  v8f acc[4];
#pragma unroll
  for (int j = 0; j < 4; ++j) acc[j] = zero8();

  for (int k0 = 0; k0 < 1024; k0 += 32) {
    {  // stage F 128x32 (async b128 x2 per thread)
      const int r = tid >> 1, cb = (tid & 1) * 16;
      const __bf16* gf = F + (size_t)r * 1024 + k0 + cb;
      cp16_g2l(gf, &lF[r * LD + cb]);
      cp16_g2l(gf + 8, &lF[r * LD + cb + 8]);
    }
    {  // stage X 32x64 transposed -> [dk][l]
      const int kk = tid >> 3, db = (tid & 7) * 8;
      const __bf16* gx = X + (size_t)(k0 + kk) * D_ + db;
#pragma unroll
      for (int e = 0; e < 8; ++e) lXt[(db + e) * LD + kk] = gx[e];
    }
    async_fence();
    __syncthreads();
    v16bf af = frag_a(&lF[(w * 16) * LD], LD);
#pragma unroll
    for (int j = 0; j < 4; ++j) {
      v16bf bf = frag_b_t(&lXt[(j * 16) * LD], LD);
      acc[j] = wmma_bf16(af, bf, acc[j]);
    }
    __syncthreads();
  }
#pragma unroll
  for (int j = 0; j < 4; ++j)
    store_acc<__bf16>(out + (size_t)(w * 16) * 64 + j * 16, 64, acc[j]);
}

// ---------------------------------------------------------------------------
// Spectral attention, one (b,h) per block, all in LDS (64 KB).
//   attn(64x64) = Qr@Kr^T + Qi@Ki^T  (K=128 over stacked [real;imag] blocks)
//   softmax rows (fp32), then [Or;Oi] = attn @ [Vr;Vi]
// Bulk tiles staged with async global->LDS.
// lab (bf16 softmaxed attn) aliases the q tile, which is dead after step 1.
// ---------------------------------------------------------------------------

__device__ __forceinline__ v16bf frag_attnA(const __bf16* lds, int m0, int k0) {
  const int lane = threadIdx.x & 31;
  const int r  = lane & 15;
  const int kh = (lane >> 4) << 4;
  const __bf16* p = lds + (k0 >> 6) * 4096 + (m0 + r) * 64 + (k0 & 63);
  v16bf f;
#pragma unroll
  for (int e = 0; e < 16; ++e) f[e] = p[kh + e];
  return f;
}

__device__ __forceinline__ v16bf frag_attnB(const __bf16* lds, int n0, int k0) {
  const int lane = threadIdx.x & 31;
  const int n    = lane & 15;
  const int koff = (lane >> 4) << 4;
  const __bf16* p = lds + (k0 >> 6) * 4096 + (n0 + n) * 64 + (k0 & 63) + koff;
  v16bf f;
#pragma unroll
  for (int e = 0; e < 16; ++e) f[e] = p[e];
  return f;
}

__global__ __launch_bounds__(256) void attn_kernel(
    const __bf16* __restrict__ ftq, const __bf16* __restrict__ ftk,
    const __bf16* __restrict__ ftv, __bf16* __restrict__ oft) {
  __shared__ __align__(16) char smem[65536];
  __bf16* lq    = (__bf16*)smem;              // 16 KB  [128][64] (Qr;Qi)
  __bf16* lk    = (__bf16*)(smem + 16384);    // 16 KB  [128][64] (Kr;Ki)
  __bf16* lv    = (__bf16*)(smem + 32768);    // 16 KB  [128][64] (Vr;Vi)
  float*  lattn = (float*)(smem + 49152);     // 16 KB  [64][64] fp32 scores
  __bf16* lab   = (__bf16*)smem;              // aliases lq (dead after step 1)

  const int bh = blockIdx.x, tid = threadIdx.x, w = tid >> 5;
  const size_t base = (size_t)bh * 128 * 64;

  {  // cooperative 16B granules, async where available
#pragma unroll
    for (int i = tid; i < 1024; i += 256) {
      cp16_g2l(ftq + base + (size_t)i * 8, lq + (size_t)i * 8);
      cp16_g2l(ftk + base + (size_t)i * 8, lk + (size_t)i * 8);
      cp16_g2l(ftv + base + (size_t)i * 8, lv + (size_t)i * 8);
    }
  }
  async_fence();
  __syncthreads();

  // Step 1: attn = Qcat(64x128) @ Kcat^T(128x64); wave w -> (mf, nfb..nfb+1)
  {
    const int mf = w >> 1, nfb = (w & 1) * 2;
    v8f a0 = zero8(), a1 = zero8();
#pragma unroll
    for (int k0 = 0; k0 < 128; k0 += 32) {
      v16bf af = frag_attnA(lq, mf * 16, k0);
      v16bf b0 = frag_attnB(lk, nfb * 16, k0);
      v16bf b1 = frag_attnB(lk, (nfb + 1) * 16, k0);
      a0 = wmma_bf16(af, b0, a0);
      a1 = wmma_bf16(af, b1, a1);
    }
    store_acc<float>(&lattn[(mf * 16) * 64 + nfb * 16], 64, a0);
    store_acc<float>(&lattn[(mf * 16) * 64 + (nfb + 1) * 16], 64, a1);
  }
  __syncthreads();

  // Step 2: per-row softmax (64 rows, fp32), write bf16 into lab
  if (tid < 64) {
    float mx = -3.0e38f;
#pragma unroll 4
    for (int s = 0; s < 64; ++s) mx = fmaxf(mx, lattn[tid * 64 + s]);
    float sum = 0.f;
#pragma unroll 4
    for (int s = 0; s < 64; ++s) {
      float e = __expf(lattn[tid * 64 + s] - mx);
      lattn[tid * 64 + s] = e;
      sum += e;
    }
    float inv = 1.f / sum;
#pragma unroll 4
    for (int s = 0; s < 64; ++s) lab[tid * 64 + s] = (__bf16)(lattn[tid * 64 + s] * inv);
  }
  __syncthreads();

  // Step 3: [Or;Oi] = attn @ [Vr;Vi]; 32 tiles / 8 waves = 4 tiles each
#pragma unroll
  for (int t = 0; t < 4; ++t) {
    const int id = w * 4 + t;
    const int oh = id >> 4;            // 0 = real, 1 = imag
    const int mf = (id >> 2) & 3;
    const int nf = id & 3;
    v8f acc = zero8();
#pragma unroll
    for (int k0 = 0; k0 < 64; k0 += 32) {
      v16bf af = frag_a(&lab[(mf * 16) * 64 + k0], 64);
      v16bf bf = frag_b_rm(&lv[(size_t)(oh * 64 + k0) * 64 + nf * 16], 64);
      acc = wmma_bf16(af, bf, acc);
    }
    store_acc<__bf16>(oft + base + (size_t)(oh * 64 + mf * 16) * 64 + nf * 16, 64, acc);
  }
}

// ---------------------------------------------------------------------------
// Truncated irDFT: per (mt, bh): out(128x64) = G[mt](128x128) @ Ocat(128x64),
// scattered into (B*L, D) layout.  G staged with async global->LDS.
// ---------------------------------------------------------------------------
__global__ __launch_bounds__(256) void idft_kernel(
    const __bf16* __restrict__ G, const __bf16* __restrict__ oft,
    __bf16* __restrict__ outm) {
  const int LD = 40;
  __shared__ __align__(16) __bf16 lG[128 * LD];
  __shared__ __align__(16) __bf16 lBt[64 * LD];
  const int mt = blockIdx.x;          // 0..7, rows mt*128..+127 of L
  const int bh = blockIdx.y;
  const int b = bh >> 4, h = bh & 15;
  const __bf16* Bp = oft + (size_t)bh * 128 * 64;
  __bf16* C = outm + (size_t)b * L_ * D_ + (size_t)mt * 128 * D_ + h * 64;
  const int tid = threadIdx.x, w = tid >> 5;

  v8f acc[4];
#pragma unroll
  for (int j = 0; j < 4; ++j) acc[j] = zero8();

#pragma unroll
  for (int k0 = 0; k0 < 128; k0 += 32) {
    {  // stage G 128x32 (async b128 x2 per thread)
      const int r = tid >> 1, cb = (tid & 1) * 16;
      const __bf16* gg = G + (size_t)(mt * 128 + r) * 128 + k0 + cb;
      cp16_g2l(gg, &lG[r * LD + cb]);
      cp16_g2l(gg + 8, &lG[r * LD + cb + 8]);
    }
    {  // stage O 32x64 transposed -> [dk][k]
      const int kk = tid >> 3, db = (tid & 7) * 8;
      const __bf16* gb = Bp + (size_t)(k0 + kk) * 64 + db;
#pragma unroll
      for (int e = 0; e < 8; ++e) lBt[(db + e) * LD + kk] = gb[e];
    }
    async_fence();
    __syncthreads();
    v16bf af = frag_a(&lG[(w * 16) * LD], LD);
#pragma unroll
    for (int j = 0; j < 4; ++j) {
      v16bf bf = frag_b_t(&lBt[(j * 16) * LD], LD);
      acc[j] = wmma_bf16(af, bf, acc[j]);
    }
    __syncthreads();
  }
#pragma unroll
  for (int j = 0; j < 4; ++j)
    store_acc<__bf16>(C + (size_t)(w * 16) * D_ + j * 16, D_, acc[j]);
}

// ---------------------------------------------------------------------------
// Host-side orchestration
// ---------------------------------------------------------------------------
extern "C" void kernel_launch(void* const* d_in, const int* in_sizes, int n_in,
                              void* d_out, int out_size, void* d_ws, size_t ws_size,
                              hipStream_t stream) {
  const float* queries = (const float*)d_in[0];
  const float* keys    = (const float*)d_in[1];
  const float* values  = (const float*)d_in[2];
  const float* Wq = (const float*)d_in[3];  const float* bq = (const float*)d_in[4];
  const float* Wk = (const float*)d_in[5];  const float* bk = (const float*)d_in[6];
  const float* Wv = (const float*)d_in[7];  const float* bv = (const float*)d_in[8];
  const float* Wo = (const float*)d_in[9];  const float* bo = (const float*)d_in[10];
  float* out = (float*)d_out;

  // workspace carve-up (bf16 everywhere)
  size_t off = 0;
  auto carve = [&](size_t bytes) {
    void* p = (char*)d_ws + off;
    off += (bytes + 255) & ~(size_t)255;
    return p;
  };
  const size_t projElems = (size_t)BL_ * D_;        // 16.78M per tensor
  const size_t ftElems   = (size_t)BH_ * 128 * 64;  // 2.10M per tensor
  __bf16* proj = (__bf16*)carve(3 * projElems * sizeof(__bf16));
  __bf16* ft   = (__bf16*)carve(3 * ftElems * sizeof(__bf16));
  __bf16* oft  = (__bf16*)carve(ftElems * sizeof(__bf16));
  __bf16* outm = (__bf16*)carve(projElems * sizeof(__bf16));
  __bf16* F    = (__bf16*)carve((size_t)128 * 1024 * sizeof(__bf16));
  __bf16* G    = (__bf16*)carve((size_t)1024 * 128 * sizeof(__bf16));

  fill_F_kernel<<<512, 256, 0, stream>>>(F);
  fill_G_kernel<<<512, 256, 0, stream>>>(G);

  dim3 gGemm(D_ / 128, BL_ / 128);  // (8, 128)
  gemm_bias_kernel<float, __bf16><<<gGemm, 256, 0, stream>>>(
      queries, Wq, bq, proj + 0 * projElems, BL_, D_, D_);
  gemm_bias_kernel<float, __bf16><<<gGemm, 256, 0, stream>>>(
      keys, Wk, bk, proj + 1 * projElems, BL_, D_, D_);
  gemm_bias_kernel<float, __bf16><<<gGemm, 256, 0, stream>>>(
      values, Wv, bv, proj + 2 * projElems, BL_, D_, D_);

  dft_kernel<<<3 * BH_, 256, 0, stream>>>(F, proj, ft);

  attn_kernel<<<BH_, 256, 0, stream>>>(ft + 0 * ftElems, ft + 1 * ftElems,
                                       ft + 2 * ftElems, oft);

  idft_kernel<<<dim3(L_ / 128, BH_), 256, 0, stream>>>(G, oft, outm);

  gemm_bias_kernel<__bf16, float><<<gGemm, 256, 0, stream>>>(
      outm, Wo, bo, out, BL_, D_, D_);
}
